// CNNGNNModel_51264729645229
// MI455X (gfx1250) — matrix operand order repaired and checked by reading, not compiled, not run on hardware
//
#include <hip/hip_runtime.h>

// ---------------------------------------------------------------------------
// GCN forward for MI455X (gfx1250, wave32, WMMA).
// Dense math in bf16 WMMA (f32 accum); graph scatter via L2-resident f32
// atomics; norm via degree atomics + rsqrt.
// ---------------------------------------------------------------------------

typedef __attribute__((ext_vector_type(16))) __bf16 v16bf;
typedef __attribute__((ext_vector_type(8)))  __bf16 v8bf;
typedef __attribute__((ext_vector_type(8)))  float  v8f;

static __device__ __forceinline__ unsigned short f32_to_bf16_bits(float f) {
  unsigned u = __builtin_bit_cast(unsigned, f);
  unsigned r = u + 0x7FFFu + ((u >> 16) & 1u);   // round-to-nearest-even
  return (unsigned short)(r >> 16);
}

// ------------------------- graph / elementwise -----------------------------

__global__ void k_fill_f32(float* __restrict__ p, float v, int n) {
  int i = blockIdx.x * blockDim.x + threadIdx.x;
  if (i < n) p[i] = v;
}

__global__ void k_deg_accum(const long long* __restrict__ col,
                            float* __restrict__ deg, int e) {
  int stride = gridDim.x * blockDim.x;
  for (int i = blockIdx.x * blockDim.x + threadIdx.x; i < e; i += stride)
    atomicAdd(&deg[(int)col[i]], 1.0f);
}

__global__ void k_rsqrt(const float* __restrict__ deg,
                        float* __restrict__ dinv, int n) {
  int i = blockIdx.x * blockDim.x + threadIdx.x;
  if (i < n) { float d = deg[i]; dinv[i] = d > 0.f ? rsqrtf(d) : 0.f; }
}

// W [K,N] f32 row-major  ->  Wt [N,K] bf16 row-major (K contiguous per row)
__global__ void k_transpose_bf16(const float* __restrict__ W,
                                 unsigned short* __restrict__ Wt, int K, int N) {
  int i = blockIdx.x * blockDim.x + threadIdx.x;
  if (i < K * N) {
    int k = i / N, n = i % N;
    Wt[(size_t)n * K + k] = f32_to_bf16_bits(W[i]);
  }
}

// AGG[i][:] = H[i][:] * dinv[i]^2   (self-loop term; also initializes AGG)
__global__ void k_selfloop_init(const float* __restrict__ H,
                                const float* __restrict__ dinv,
                                float* __restrict__ AGG, int nD4, int D) {
  int i = blockIdx.x * blockDim.x + threadIdx.x;
  if (i < nD4) {
    int node = (i * 4) / D;
    float s = dinv[node]; s = s * s;
    float4 h = ((const float4*)H)[i];
    float4 o; o.x = h.x * s; o.y = h.y * s; o.z = h.z * s; o.w = h.w * s;
    ((float4*)AGG)[i] = o;
  }
}

// AGG[col[e]][:] += H[row[e]][:] * dinv[row]*dinv[col]  (L2-resident atomics)
__global__ void k_edge_scatter(const long long* __restrict__ row,
                               const long long* __restrict__ col,
                               const float* __restrict__ dinv,
                               const float* __restrict__ H,
                               float* __restrict__ AGG, int E, int D) {
  const int CH = D / 4;                       // float4 chunks per edge
  size_t total  = (size_t)E * CH;
  size_t stride = (size_t)gridDim.x * blockDim.x;
  for (size_t w = (size_t)blockIdx.x * blockDim.x + threadIdx.x; w < total;
       w += stride) {
    size_t e = w / CH;
    int    c = (int)(w % CH);
    int r = (int)row[e], t = (int)col[e];
    float nrm = dinv[r] * dinv[t];
    float4 h = *(const float4*)(H + (size_t)r * D + c * 4);
    float* dst = AGG + (size_t)t * D + c * 4;
    atomicAdd(dst + 0, h.x * nrm);
    atomicAdd(dst + 1, h.y * nrm);
    atomicAdd(dst + 2, h.z * nrm);
    atomicAdd(dst + 3, h.w * nrm);
  }
}

// out = bf16(relu(AGG + b))  packed for the next WMMA GEMM's A operand
__global__ void k_bias_relu_bf16(const float* __restrict__ AGG,
                                 const float* __restrict__ b,
                                 unsigned short* __restrict__ out,
                                 int nD4, int D) {
  int i = blockIdx.x * blockDim.x + threadIdx.x;
  if (i < nD4) {
    int d = (i * 4) % D;
    float4 a  = ((const float4*)AGG)[i];
    float4 bb = *(const float4*)(b + d);
    uint2 p;
    p.x = (unsigned)f32_to_bf16_bits(fmaxf(a.x + bb.x, 0.f)) |
          ((unsigned)f32_to_bf16_bits(fmaxf(a.y + bb.y, 0.f)) << 16);
    p.y = (unsigned)f32_to_bf16_bits(fmaxf(a.z + bb.z, 0.f)) |
          ((unsigned)f32_to_bf16_bits(fmaxf(a.w + bb.w, 0.f)) << 16);
    ((uint2*)out)[i] = p;
  }
}

// ------------------------------ WMMA GEMM ----------------------------------
// C[M,N] = A[M,K] @ Bt[N,K]^T (+ bias).  A is f32 (converted on the fly) or
// raw-bf16.  Block: 256 threads = 8 waves (2 M-groups x 4 N-groups).
// Block tile 128x128, wave tile 64x32 (4x2 WMMA accs), K-step 32.
template <bool A_F32, bool HAS_BIAS>
__global__ void __launch_bounds__(256) k_gemm_wmma(
    const void* __restrict__ Av, const unsigned short* __restrict__ Bt,
    const float* __restrict__ bias, float* __restrict__ C,
    int M, int N, int K) {
  constexpr int BM = 128, BN = 128, BK = 32;
  constexpr int LDA = BK + 8;                 // 80-byte LDS rows (conflict pad)
  __shared__ unsigned short sA[BM * LDA];

  const int tid  = threadIdx.x;
  const int wave = tid >> 5;
  const int lane = tid & 31;
  const int wM   = wave >> 2;                 // 0..1
  const int wN   = wave & 3;                  // 0..3
  const int l15  = lane & 15;
  const int lh   = lane >> 4;                 // half-wave K/M group select

  const int mBlock = blockIdx.y * BM;
  const int nBlock = blockIdx.x * BN;

  v8f acc[4][2] = {};

  for (int k0 = 0; k0 < K; k0 += BK) {
    // ---- stage A tile (BM x BK) into LDS as bf16 ----
    if constexpr (A_F32) {
      const float* A = (const float*)Av;
      int r = tid >> 3;                       // 0..31
      int c = tid & 7;                        // float4 chunk
#pragma unroll
      for (int rr = 0; rr < 4; ++rr) {
        int row  = r + rr * 32;
        int grow = mBlock + row; if (grow >= M) grow = M - 1;  // clamp (EXEC stays full)
        const float* src = A + (size_t)grow * K + k0 + c * 4;
        if (k0 + BK < K) __builtin_prefetch(src + BK, 0, 3);   // global_prefetch
        float4 f = *(const float4*)src;
        uint2 p;
        p.x = (unsigned)f32_to_bf16_bits(f.x) |
              ((unsigned)f32_to_bf16_bits(f.y) << 16);
        p.y = (unsigned)f32_to_bf16_bits(f.z) |
              ((unsigned)f32_to_bf16_bits(f.w) << 16);
        *(uint2*)&sA[row * LDA + c * 4] = p;
      }
    } else {
      const unsigned short* A = (const unsigned short*)Av;
      int row  = tid >> 1;                    // 0..127
      int c    = tid & 1;                     // 16-half chunk
      int grow = mBlock + row; if (grow >= M) grow = M - 1;
      const uint4* src = (const uint4*)(A + (size_t)grow * K + k0 + c * 16);
      *(uint4*)&sA[row * LDA + c * 16]     = src[0];
      *(uint4*)&sA[row * LDA + c * 16 + 8] = src[1];
    }
    __syncthreads();

    // ---- B fragments straight from L2-resident transposed weights ----
    // B layout used here (32x16 bf16): lane = column, half-wave selects K 0..15/16..31,
    // K contiguous within lane  ->  one 32-byte load per fragment.
    v16bf bfr[2];
#pragma unroll
    for (int nt = 0; nt < 2; ++nt) {
      int colc = nBlock + wN * 32 + nt * 16 + l15;
      if (colc >= N) colc = N - 1;            // clamp, store is guarded later
      bfr[nt] = *(const v16bf*)(Bt + (size_t)colc * K + k0 + lh * 16);
    }

    // ---- 4x2 WMMA per wave ----
    // A layout used here (16x32 bf16): lane%16 = row; lane/16 selects K 0..15/16..31,
    // K contiguous within lane  ->  two 16-byte LDS loads.
#pragma unroll
    for (int mt = 0; mt < 4; ++mt) {
      const unsigned short* ab =
          &sA[(wM * 64 + mt * 16 + l15) * LDA + lh * 16];
      v8bf alo = *(const v8bf*)(ab);
      v8bf ahi = *(const v8bf*)(ab + 8);
      v16bf af = __builtin_shufflevector(alo, ahi, 0, 1, 2, 3, 4, 5, 6, 7, 8,
                                         9, 10, 11, 12, 13, 14, 15);
#pragma unroll
      for (int nt = 0; nt < 2; ++nt)
        acc[mt][nt] = __builtin_amdgcn_wmma_f32_16x16x32_bf16(
            false, af, false, bfr[nt], (short)0, acc[mt][nt], false, false);
    }
    __syncthreads();
  }

  // ---- epilogue: C layout VGPR r -> M = r + 8*(lane/16), N = lane%16 ----
#pragma unroll
  for (int nt = 0; nt < 2; ++nt) {
    int colc = nBlock + wN * 32 + nt * 16 + l15;
    if (colc < N) {
      float bv = HAS_BIAS ? bias[colc] : 0.0f;
#pragma unroll
      for (int mt = 0; mt < 4; ++mt) {
#pragma unroll
        for (int r = 0; r < 8; ++r) {
          int rowc = mBlock + wM * 64 + mt * 16 + lh * 8 + r;
          if (rowc < M) C[(size_t)rowc * N + colc] = acc[mt][nt][r] + bv;
        }
      }
    }
  }
}

// ------------------------------- driver ------------------------------------

extern "C" void kernel_launch(void* const* d_in, const int* in_sizes, int n_in,
                              void* d_out, int out_size, void* d_ws,
                              size_t ws_size, hipStream_t stream) {
  (void)in_sizes; (void)n_in; (void)out_size; (void)ws_size;
  constexpr int N = 100000, E = 3200000, IN = 512, HID = 256, CLS = 1000;

  const float*     x   = (const float*)d_in[0];
  const long long* ei  = (const long long*)d_in[1];   // [2, E] int64
  const float*     W1  = (const float*)d_in[2];
  const float*     b1  = (const float*)d_in[3];
  const float*     W2  = (const float*)d_in[4];
  const float*     b2  = (const float*)d_in[5];
  const float*     Wfc = (const float*)d_in[6];
  const float*     bfc = (const float*)d_in[7];
  float*           out = (float*)d_out;
  const long long* erow = ei;                          // sources
  const long long* ecol = ei + E;                      // targets

  // workspace layout (~258 MB)
  char* w = (char*)d_ws;
  auto take = [&](size_t bytes) {
    char* p = w; w += (bytes + 255) & ~(size_t)255; return p;
  };
  float*          deg  = (float*)take(sizeof(float) * N);
  float*          dinv = (float*)take(sizeof(float) * N);
  unsigned short* W1t  = (unsigned short*)take(2ull * HID * IN);
  unsigned short* W2t  = (unsigned short*)take(2ull * HID * HID);
  unsigned short* Wfct = (unsigned short*)take(2ull * CLS * HID);
  float*          H    = (float*)take(4ull * N * HID);
  float*          AGG  = (float*)take(4ull * N * HID);
  unsigned short* HB   = (unsigned short*)take(2ull * N * HID);

  const int T = 256;
  // --- GCN normalization: deg = 1 (self loop) + in-degree; dinv = rsqrt ---
  k_fill_f32<<<(N + T - 1) / T, T, 0, stream>>>(deg, 1.0f, N);
  k_deg_accum<<<4096, T, 0, stream>>>(ecol, deg, E);
  k_rsqrt<<<(N + T - 1) / T, T, 0, stream>>>(deg, dinv, N);
  // --- weights -> bf16, transposed to [out, in] for K-contiguous B frags ---
  k_transpose_bf16<<<(IN * HID + T - 1) / T, T, 0, stream>>>(W1, W1t, IN, HID);
  k_transpose_bf16<<<(HID * HID + T - 1) / T, T, 0, stream>>>(W2, W2t, HID, HID);
  k_transpose_bf16<<<(HID * CLS + T - 1) / T, T, 0, stream>>>(Wfc, Wfct, HID, CLS);

  dim3 blk(T);
  dim3 g12(HID / 128, (N + 127) / 128);
  const int nD4 = N * HID / 4;

  // --- layer 1: H = x @ W1 ; AGG = norm-scatter(H) ; HB = bf16(relu(+b1)) ---
  k_gemm_wmma<true, false><<<g12, blk, 0, stream>>>(x, W1t, nullptr, H, N, HID, IN);
  k_selfloop_init<<<(nD4 + T - 1) / T, T, 0, stream>>>(H, dinv, AGG, nD4, HID);
  k_edge_scatter<<<8192, T, 0, stream>>>(erow, ecol, dinv, H, AGG, E, HID);
  k_bias_relu_bf16<<<(nD4 + T - 1) / T, T, 0, stream>>>(AGG, b1, HB, nD4, HID);

  // --- layer 2 ---
  k_gemm_wmma<false, false><<<g12, blk, 0, stream>>>(HB, W2t, nullptr, H, N, HID, HID);
  k_selfloop_init<<<(nD4 + T - 1) / T, T, 0, stream>>>(H, dinv, AGG, nD4, HID);
  k_edge_scatter<<<8192, T, 0, stream>>>(erow, ecol, dinv, H, AGG, E, HID);
  k_bias_relu_bf16<<<(nD4 + T - 1) / T, T, 0, stream>>>(AGG, b2, HB, nD4, HID);

  // --- classifier: out = HB @ Wfc + bfc  (fp32 logits to d_out) ---
  dim3 g3((CLS + 127) / 128, (N + 127) / 128);
  k_gemm_wmma<false, true><<<g3, blk, 0, stream>>>(HB, Wfct, bfc, out, N, CLS, HID);
}
